// QuantLinear_54150947668071
// MI455X (gfx1250) — matrix operand order repaired and checked by reading, not compiled, not run on hardware
//
#include <hip/hip_runtime.h>

// GEMM dims from the reference: M = B*S = 8192, N = DOUT = 4096, K = DIN = 4096
#define GEMM_M 8192
#define GEMM_N 4096
#define GEMM_K 4096

// ---- main bf16 GEMM tiling ----
#define BM 256      // block tile M (4 waves x 64)
#define BN 128      // block tile N (2 waves x 64)
#define BK 32       // K chunk per LDS stage (one WMMA K-step)
#define LDT 40      // LDS row stride in bf16 elements (32 + 8 pad)
#define THREADS 256 // 8 waves (wave32)

typedef __attribute__((ext_vector_type(16))) __bf16         v16bf;
typedef __attribute__((ext_vector_type(8)))  float          v8f;
typedef __attribute__((ext_vector_type(8)))  unsigned short ush8;
typedef __attribute__((ext_vector_type(4)))  int            v4i;

union Frag {
    v16bf v;
    ush8  h[2];
};

__device__ __forceinline__ unsigned short f2bf(float f) {
    // round-to-nearest-even fp32 -> bf16 (exact for int8-valued floats)
    unsigned int u = __float_as_uint(f);
    u += 0x7FFFu + ((u >> 16) & 1u);
    return (unsigned short)(u >> 16);
}

// ---- CDNA5 async global->LDS path ----
#if defined(__has_builtin)
#  if __has_builtin(__builtin_amdgcn_global_load_async_to_lds_b128)
#    define HAVE_ASYNC_LDS 1
#  endif
#endif

#define AS_GLOBAL __attribute__((address_space(1)))
#define AS_LDS    __attribute__((address_space(3)))

#ifdef HAVE_ASYNC_LDS
__device__ __forceinline__ void async_cp_b128(const void* g, void* l) {
    __builtin_amdgcn_global_load_async_to_lds_b128(
        (AS_GLOBAL v4i*)(g), (AS_LDS v4i*)(l), 0, 0);
}
__device__ __forceinline__ void wait_async0() {
#if defined(__has_builtin) && __has_builtin(__builtin_amdgcn_s_wait_asynccnt)
    __builtin_amdgcn_s_wait_asynccnt(0);
#else
    asm volatile("s_wait_asynccnt 0x0" ::: "memory");
#endif
}
#endif

// =====================================================================
// Pre-pass 1: W int8 -> bf16 (exact).  One int4 (16 int8) per thread.
// =====================================================================
__global__ __launch_bounds__(256)
void convert_w_bf16(const signed char* __restrict__ W, unsigned short* __restrict__ Wb) {
    int idx = blockIdx.x * 256 + threadIdx.x;      // int4 index
    int4 v = ((const int4*)W)[idx];
    const signed char* pc = (const signed char*)&v;
    unsigned int pk[8];
#pragma unroll
    for (int j = 0; j < 8; ++j)
        pk[j] = (unsigned int)f2bf((float)pc[2 * j]) |
                ((unsigned int)f2bf((float)pc[2 * j + 1]) << 16);
    uint4* dst = (uint4*)(Wb + (size_t)idx * 16);
    dst[0] = make_uint4(pk[0], pk[1], pk[2], pk[3]);
    dst[1] = make_uint4(pk[4], pk[5], pk[6], pk[7]);
}

// =====================================================================
// Pre-pass 2: X fp32 -> bf16 (RNE).  8 floats per thread.
// =====================================================================
__global__ __launch_bounds__(256)
void convert_x_bf16(const float* __restrict__ X, unsigned short* __restrict__ Xb) {
    size_t idx = (size_t)(blockIdx.x * 256 + threadIdx.x) * 8;
    float4 a = *(const float4*)(X + idx);
    float4 b = *(const float4*)(X + idx + 4);
    uint4 o;
    o.x = (unsigned int)f2bf(a.x) | ((unsigned int)f2bf(a.y) << 16);
    o.y = (unsigned int)f2bf(a.z) | ((unsigned int)f2bf(a.w) << 16);
    o.z = (unsigned int)f2bf(b.x) | ((unsigned int)f2bf(b.y) << 16);
    o.w = (unsigned int)f2bf(b.z) | ((unsigned int)f2bf(b.w) << 16);
    *(uint4*)(Xb + idx) = o;
}

// =====================================================================
// Main GEMM: pure bf16 inputs, WMMA f32 accumulation, fused scale+bias.
// Block tile 256x128, 8 waves in 4(M)x2(N), wave tile 64x64.
// K-loop kept at unroll 1 so only one iteration's fragments are live:
//   acc 128 + A-frags 32 + B-frag 8 VGPRs  ->  stays under 256 VGPRs.
// =====================================================================
__global__ __launch_bounds__(THREADS)
void qlinear_gemm_bf16(const unsigned short* __restrict__ Xb,  // [M,K] bf16
                       const unsigned short* __restrict__ Wb,  // [N,K] bf16
                       const float* __restrict__ scale_p,
                       const float* __restrict__ bias,
                       float* __restrict__ out)
{
    __shared__ __align__(16) unsigned short As[2][BM * LDT];   // 40 KB
    __shared__ __align__(16) unsigned short Bs[2][BN * LDT];   // 20 KB

    const int tid  = threadIdx.x;
    const int lane = tid & 31;
    const int wave = tid >> 5;
    const int lm   = lane & 15;
    const int hl   = lane >> 4;

    const int bn0 = blockIdx.x * BN;
    const int bm0 = blockIdx.y * BM;
    const int wm  = (wave & 3) * 64;
    const int wn  = (wave >> 2) * 64;

    // Per-thread fragment base offsets (in bf16 elements), loop-invariant.
    const int aBase = (wm + lm) * LDT + hl * 8;   // + mi*16*LDT, +16 for 2nd half
    const int bBase = (wn + lm) * LDT + hl * 8;   // + ni*16*LDT, +16 for 2nd half

    v8f acc[4][4];
#pragma unroll
    for (int mi = 0; mi < 4; ++mi)
#pragma unroll
        for (int ni = 0; ni < 4; ++ni)
            acc[mi][ni] = (v8f){0.f, 0.f, 0.f, 0.f, 0.f, 0.f, 0.f, 0.f};

    // Tile copy geometry: 16B chunks, 4 chunks per 32-bf16 row.
    // A: 256 rows -> 1024 chunks -> 4/thread.  B: 128 rows -> 512 -> 2/thread.

#ifdef HAVE_ASYNC_LDS
    auto loadTileAsync = [&](int k0, int buf) {
#pragma unroll
        for (int r = 0; r < 4; ++r) {
            int idx = tid + r * THREADS;
            int row = idx >> 2, c8 = idx & 3;
            async_cp_b128(Xb + (size_t)(bm0 + row) * GEMM_K + k0 + c8 * 8,
                          &As[buf][row * LDT + c8 * 8]);
        }
#pragma unroll
        for (int r = 0; r < 2; ++r) {
            int idx = tid + r * THREADS;
            int row = idx >> 2, c8 = idx & 3;
            async_cp_b128(Wb + (size_t)(bn0 + row) * GEMM_K + k0 + c8 * 8,
                          &Bs[buf][row * LDT + c8 * 8]);
        }
    };
#else
    uint4 sA[4], sB[2];
    auto loadTile = [&](int k0) {
#pragma unroll
        for (int r = 0; r < 4; ++r) {
            int idx = tid + r * THREADS;
            int row = idx >> 2, c8 = idx & 3;
            sA[r] = *(const uint4*)(Xb + (size_t)(bm0 + row) * GEMM_K + k0 + c8 * 8);
        }
#pragma unroll
        for (int r = 0; r < 2; ++r) {
            int idx = tid + r * THREADS;
            int row = idx >> 2, c8 = idx & 3;
            sB[r] = *(const uint4*)(Wb + (size_t)(bn0 + row) * GEMM_K + k0 + c8 * 8);
        }
    };
    auto storeTile = [&](int buf) {
#pragma unroll
        for (int r = 0; r < 4; ++r) {
            int idx = tid + r * THREADS;
            int row = idx >> 2, c8 = idx & 3;
            *(uint4*)(&As[buf][row * LDT + c8 * 8]) = sA[r];
        }
#pragma unroll
        for (int r = 0; r < 2; ++r) {
            int idx = tid + r * THREADS;
            int row = idx >> 2, c8 = idx & 3;
            *(uint4*)(&Bs[buf][row * LDT + c8 * 8]) = sB[r];
        }
    };
#endif

    auto compute = [&](const unsigned short* Ab, const unsigned short* Bb) {
        Frag a[4];
        // A 16x32 bf16 ISA layout: lanes 0-15 hold K0-7 & K16-23; lanes 16-31 K8-15 & K24-31.
#pragma unroll
        for (int mi = 0; mi < 4; ++mi) {
            const unsigned short* p = Ab + aBase + mi * (16 * LDT);
            a[mi].h[0] = *(const ush8*)(p);
            a[mi].h[1] = *(const ush8*)(p + 16);
        }
        // B 32x16 layout: lane = column n (weight row); lanes 0-15 K0-7 & K16-23; lanes 16-31 K8-15 & K24-31.
#pragma unroll
        for (int ni = 0; ni < 4; ++ni) {
            Frag b;
            const unsigned short* p = Bb + bBase + ni * (16 * LDT);
            b.h[0] = *(const ush8*)(p);
            b.h[1] = *(const ush8*)(p + 16);
#pragma unroll
            for (int mi = 0; mi < 4; ++mi)
                acc[mi][ni] = __builtin_amdgcn_wmma_f32_16x16x32_bf16(
                    false, a[mi].v, false, b.v, (short)0, acc[mi][ni], false, false);
        }
    };

    const int KT = GEMM_K / BK;  // 128

#ifdef HAVE_ASYNC_LDS
    loadTileAsync(0, 0);
    wait_async0();
    __syncthreads();
#pragma unroll 1
    for (int kt = 0; kt < KT; ++kt) {
        const int cur = kt & 1;
        if (kt + 1 < KT) loadTileAsync((kt + 1) * BK, cur ^ 1); // DMA under compute
        compute(As[cur], Bs[cur]);
        wait_async0();
        __syncthreads();      // single barrier per K-step
    }
#else
    loadTile(0);
    storeTile(0);
    __syncthreads();
#pragma unroll 1
    for (int kt = 0; kt < KT; ++kt) {
        const int cur = kt & 1;
        if (kt + 1 < KT) loadTile((kt + 1) * BK);
        compute(As[cur], Bs[cur]);
        __syncthreads();
        if (kt + 1 < KT) storeTile(cur ^ 1);
        __syncthreads();
    }
#endif

    // Epilogue: fp32 accum * scale + bias
    const float s = scale_p[0];
#pragma unroll
    for (int mi = 0; mi < 4; ++mi) {
#pragma unroll
        for (int ni = 0; ni < 4; ++ni) {
            int   col = bn0 + wn + ni * 16 + lm;
            float bz  = bias[col];
#pragma unroll
            for (int r = 0; r < 8; ++r) {
                int row = bm0 + wm + mi * 16 + r + hl * 8;
                out[(size_t)row * GEMM_N + col] = acc[mi][ni][r] * s + bz;
            }
        }
    }
}

// =====================================================================
// Fallback fused kernel (conversion in-loop) for when d_ws is too small.
// =====================================================================
#define FBM 128
#define FBK 32
#define FLDT 40

__global__ __launch_bounds__(THREADS)
void qlinear_fused(const float* __restrict__ X, const signed char* __restrict__ W,
                   const float* __restrict__ scale_p, const float* __restrict__ bias,
                   float* __restrict__ out)
{
    __shared__ __align__(16) unsigned short As[2][FBM * FLDT];
    __shared__ __align__(16) unsigned short Bsm[2][FBM * FLDT];

    const int tid = threadIdx.x, lane = tid & 31, wave = tid >> 5;
    const int lm = lane & 15, hl = lane >> 4;
    const int bn0 = blockIdx.x * FBM, bm0 = blockIdx.y * FBM;
    const int wm = (wave & 3) * 32, wn = (wave >> 2) * 64;

    v8f acc[2][4];
#pragma unroll
    for (int mi = 0; mi < 2; ++mi)
#pragma unroll
        for (int ni = 0; ni < 4; ++ni)
            acc[mi][ni] = (v8f){0.f, 0.f, 0.f, 0.f, 0.f, 0.f, 0.f, 0.f};

    float4 aReg[4];
    int4   bReg;
    auto loadTile = [&](int k0) {
#pragma unroll
        for (int r = 0; r < 4; ++r) {
            int idx = tid + r * THREADS, row = idx >> 3, c4 = idx & 7;
            aReg[r] = *(const float4*)(X + (size_t)(bm0 + row) * GEMM_K + k0 + c4 * 4);
        }
        int row = tid >> 1, c16 = tid & 1;
        bReg = *(const int4*)(W + (size_t)(bn0 + row) * GEMM_K + k0 + c16 * 16);
    };
    auto storeTile = [&](int buf) {
#pragma unroll
        for (int r = 0; r < 4; ++r) {
            int idx = tid + r * THREADS, row = idx >> 3, c4 = idx & 7;
            unsigned int p0 = (unsigned int)f2bf(aReg[r].x) | ((unsigned int)f2bf(aReg[r].y) << 16);
            unsigned int p1 = (unsigned int)f2bf(aReg[r].z) | ((unsigned int)f2bf(aReg[r].w) << 16);
            *(uint2*)(&As[buf][row * FLDT + c4 * 4]) = make_uint2(p0, p1);
        }
        int row = tid >> 1, c16 = tid & 1;
        const signed char* pc = (const signed char*)&bReg;
        unsigned int pk[8];
#pragma unroll
        for (int j = 0; j < 8; ++j)
            pk[j] = (unsigned int)f2bf((float)pc[2 * j]) |
                    ((unsigned int)f2bf((float)pc[2 * j + 1]) << 16);
        uint4* dst = (uint4*)(&Bsm[buf][row * FLDT + c16 * 16]);
        dst[0] = make_uint4(pk[0], pk[1], pk[2], pk[3]);
        dst[1] = make_uint4(pk[4], pk[5], pk[6], pk[7]);
    };
    auto compute = [&](const unsigned short* Ab, const unsigned short* Bb) {
        Frag a[2], b[4];
#pragma unroll
        for (int mi = 0; mi < 2; ++mi) {
            int off = (wm + mi * 16 + lm) * FLDT + hl * 8;
            a[mi].h[0] = *(const ush8*)(Ab + off);
            a[mi].h[1] = *(const ush8*)(Ab + off + 16);
        }
#pragma unroll
        for (int ni = 0; ni < 4; ++ni) {
            int off = (wn + ni * 16 + lm) * FLDT + hl * 8;
            b[ni].h[0] = *(const ush8*)(Bb + off);
            b[ni].h[1] = *(const ush8*)(Bb + off + 16);
        }
#pragma unroll
        for (int mi = 0; mi < 2; ++mi)
#pragma unroll
            for (int ni = 0; ni < 4; ++ni)
                acc[mi][ni] = __builtin_amdgcn_wmma_f32_16x16x32_bf16(
                    false, a[mi].v, false, b[ni].v, (short)0, acc[mi][ni], false, false);
    };

    loadTile(0);
    storeTile(0);
    __syncthreads();
    const int KT = GEMM_K / FBK;
#pragma unroll 1
    for (int kt = 0; kt < KT; ++kt) {
        const int cur = kt & 1;
        if (kt + 1 < KT) loadTile((kt + 1) * FBK);
        compute(As[cur], Bsm[cur]);
        __syncthreads();
        if (kt + 1 < KT) storeTile(cur ^ 1);
        __syncthreads();
    }

    const float s = scale_p[0];
#pragma unroll
    for (int mi = 0; mi < 2; ++mi)
#pragma unroll
        for (int ni = 0; ni < 4; ++ni) {
            int   col = bn0 + wn + ni * 16 + lm;
            float bz  = bias[col];
#pragma unroll
            for (int r = 0; r < 8; ++r) {
                int row = bm0 + wm + mi * 16 + r + hl * 8;
                out[(size_t)row * GEMM_N + col] = acc[mi][ni][r] * s + bz;
            }
        }
}

// =====================================================================
extern "C" void kernel_launch(void* const* d_in, const int* in_sizes, int n_in,
                              void* d_out, int out_size, void* d_ws, size_t ws_size,
                              hipStream_t stream) {
    (void)in_sizes; (void)n_in; (void)out_size;
    const float*       x     = (const float*)d_in[0];
    const signed char* w     = (const signed char*)d_in[1];
    const float*       scale = (const float*)d_in[2];
    const float*       bias  = (const float*)d_in[3];
    float*             out   = (float*)d_out;

    const size_t wBytes = (size_t)GEMM_N * GEMM_K * sizeof(unsigned short); // 32 MB
    const size_t xBytes = (size_t)GEMM_M * GEMM_K * sizeof(unsigned short); // 64 MB

    if (ws_size >= wBytes + xBytes) {
        unsigned short* Wb = (unsigned short*)d_ws;
        unsigned short* Xb = (unsigned short*)((char*)d_ws + wBytes);
        convert_w_bf16<<<(GEMM_N * GEMM_K / 16) / 256, 256, 0, stream>>>(w, Wb);
        convert_x_bf16<<<(GEMM_M * GEMM_K / 8) / 256, 256, 0, stream>>>(x, Xb);
        dim3 grid(GEMM_N / BN, GEMM_M / BM);  // 32 x 32
        qlinear_gemm_bf16<<<grid, THREADS, 0, stream>>>(Xb, Wb, scale, bias, out);
    } else {
        dim3 grid(GEMM_N / FBM, GEMM_M / FBM);  // 32 x 64
        qlinear_fused<<<grid, THREADS, 0, stream>>>(x, w, scale, bias, out);
    }
}
